// MRSTAN_submodule_67869073211737
// MI455X (gfx1250) — matrix operand order repaired and checked by reading, not compiled, not run on hardware
//
#include <hip/hip_runtime.h>
#include <hip/hip_bf16.h>
#include <math.h>

// ---------------------------------------------------------------------------
// MRSTAN forward for MI455X (gfx1250, wave32).
// Heavy GEMMs (S = Vs @ sigmoid(...), tmp_k = (cheb*sAt)^T @ X) run on the
// matrix path via v_wmma_f32_16x16x32_f16 (f16 in, f32 accumulate, K=512),
// 128x64 workgroup tiles, 2x2 WMMAs per wave per K-step, register-staged
// software pipelining (global loads for K+1 overlap WMMAs on K).
//
// Input flattening assumption (setup_inputs() insertion order):
//  0:x  1..10: block0 {W1,W2,W3,bs,Vs,Theta,Wres,bres,gamma,beta}
//  11..20: block1 (same order)  21:Wfc 22:bfc 23:Wg1 24:bg1 25:Wg2 26:bg2
//  27:Wfl 28:bfl 29:Ww 30:bw 31:Wd 32:bd 33:cheb
// ---------------------------------------------------------------------------

#define Bb 32
#define Nn 512
#define Tt 12
#define Cc 64
#define NN2 (512 * 512)

typedef __attribute__((ext_vector_type(16))) _Float16 v16h;
typedef __attribute__((ext_vector_type(8)))  float    v8f;

union FragU { v16h v; unsigned u[8]; };

// A-matrix 16-bit 16x32 layout (ISA 7.12.2): lane<16 -> M=lane, VGPR j holds
// K pairs {2j,2j+1} for j<4 and {16+2(j-4),...} for j>=4; lanes>=16 add K+8.
__device__ __forceinline__ v16h frag_a(const _Float16* rowbase, int lane) {
  FragU f;
  const int hi8 = (lane & 16) ? 8 : 0;
#pragma unroll
  for (int j = 0; j < 8; ++j) {
    const int bk = ((j < 4) ? (2 * j) : (8 + 2 * j)) + hi8; // 0,2,4,6,16,18,20,22 (+8)
    f.u[j] = *(const unsigned*)(rowbase + bk);
  }
  return f.v;
}

// B-matrix: same K order as A (VGPR j holds K {2j,2j+1} for j<4, {16+2(j-4),...}
// for j>=4; lanes>=16 add K+8); column N = lane&15. LDS tile is stored K-transposed (Bt[n][k]).
__device__ __forceinline__ v16h frag_b(const _Float16* rowbase, int lane) {
  FragU f;
  const int koff = (lane & 16) ? 8 : 0;
#pragma unroll
  for (int j = 0; j < 8; ++j)
    f.u[j] = *(const unsigned*)(rowbase + koff + ((j < 4) ? (2 * j) : (8 + 2 * j)));
  return f.v;
}

// -------------------- small elementwise / staging kernels -------------------

__global__ void extract_kernel(const float* __restrict__ x, float* __restrict__ X0) {
  int tid = blockIdx.x * blockDim.x + threadIdx.x; // (b*512+n)*12 + t
  int t = tid % Tt;
  int bn = tid / Tt;
  X0[tid] = x[(size_t)bn * 36 + 24 + t];
}

__global__ void zero_kernel(float* __restrict__ p, int nelem) {
  int tid = blockIdx.x * blockDim.x + threadIdx.x;
  if (tid < nelem) p[tid] = 0.f;
}

template <int F>
__global__ void lr_kernel(const float* __restrict__ X, const float* __restrict__ W1,
                          const float* __restrict__ W2, const float* __restrict__ W3,
                          float* __restrict__ L, float* __restrict__ R) {
  int tid = blockIdx.x * blockDim.x + threadIdx.x; // b*512+n
  if (tid >= Bb * Nn) return;
  int b = tid >> 9, n = tid & 511;
  const float* xp = X + (size_t)tid * F * Tt;
  float lhs[F];
  for (int f = 0; f < F; ++f) {
    float s = 0.f;
#pragma unroll
    for (int t = 0; t < Tt; ++t) s += xp[f * Tt + t] * W1[t];
    lhs[f] = s;
  }
#pragma unroll
  for (int t = 0; t < Tt; ++t) {
    float s = 0.f;
    for (int f = 0; f < F; ++f) s += lhs[f] * W2[f * Tt + t];
    L[(size_t)tid * Tt + t] = s;
  }
#pragma unroll
  for (int t = 0; t < Tt; ++t) {
    float s = 0.f;
    for (int f = 0; f < F; ++f) s += W3[f] * xp[f * Tt + t];
    R[((size_t)b * Tt + t) * Nn + n] = s;
  }
}

__global__ void prod_sig_kernel(const float* __restrict__ L, const float* __restrict__ R,
                                const float* __restrict__ bs, float* __restrict__ P) {
  size_t tid = (size_t)blockIdx.x * blockDim.x + threadIdx.x; // b*NN + n*512 + m
  int m = (int)(tid & 511);
  int n = (int)((tid >> 9) & 511);
  int b = (int)(tid >> 18);
  const float* Lp = L + ((size_t)(b * Nn + n)) * Tt;
  float s = 0.f;
#pragma unroll
  for (int t = 0; t < Tt; ++t) s += Lp[t] * R[((size_t)b * Tt + t) * Nn + m];
  s += bs[n * Nn + m];
  P[tid] = 1.f / (1.f + __expf(-s));
}

// softmax over axis n (rows) for each column j; one thread per (b,j) column
__global__ void softmax_col_kernel(float* __restrict__ S) {
  int tid = blockIdx.x * blockDim.x + threadIdx.x; // b*512 + j
  int b = tid >> 9, j = tid & 511;
  float* Sb = S + (size_t)b * NN2 + j;
  float mx = -1e30f;
  for (int n = 0; n < Nn; ++n) mx = fmaxf(mx, Sb[(size_t)n * Nn]);
  float sum = 0.f;
  for (int n = 0; n < Nn; ++n) sum += __expf(Sb[(size_t)n * Nn] - mx);
  float inv = 1.f / sum;
  for (int n = 0; n < Nn; ++n) Sb[(size_t)n * Nn] = __expf(Sb[(size_t)n * Nn] - mx) * inv;
}

// -------------------------- WMMA GEMM kernels -------------------------------
// Workgroup: 256 threads = 8 waves (4 M x 2 N). WG tile 128(M) x 64(N).
// Each wave: 32x32 output = 2x2 accumulators, 4 independent WMMAs per K-step
// (covers the f16 WMMA co-execution hazard window). K advances by 32 with
// register-staged double buffering: global loads for tile k+1 are issued
// before the fragment loads / WMMAs of tile k.

__launch_bounds__(256)
__global__ void gemm_s_kernel(const float* __restrict__ Vs, const float* __restrict__ P,
                              float* __restrict__ S) {
  __shared__ _Float16 As[128][40];
  __shared__ _Float16 Bt[64][40];
  const int b = blockIdx.z;
  const int m0 = blockIdx.y * 128;
  const int n0 = blockIdx.x * 64;
  const int tid = threadIdx.x;
  const int lane = tid & 31, wave = tid >> 5;
  const int wm = (wave >> 1) * 32; // 0,32,64,96
  const int wn = (wave & 1) * 32;  // 0,32
  const float* Pb = P + (size_t)b * NN2;

  const int akk = tid & 31, amr = tid >> 5;  // A: 128x32, rows amr+8r (r<16)
  const int bnn = tid & 63, bk0 = tid >> 6;  // B: 32x64,  k = bk0+4r (r<8)

  float aReg[16], bReg[8];
#pragma unroll
  for (int r = 0; r < 16; ++r)
    aReg[r] = Vs[(size_t)(m0 + amr + 8 * r) * 512 + akk];
#pragma unroll
  for (int r = 0; r < 8; ++r)
    bReg[r] = Pb[(size_t)(bk0 + 4 * r) * 512 + n0 + bnn];

  v8f acc00 = {}, acc01 = {}, acc10 = {}, acc11 = {};
  for (int k0 = 0; k0 < 512; k0 += 32) {
#pragma unroll
    for (int r = 0; r < 16; ++r) As[amr + 8 * r][akk] = (_Float16)aReg[r];
#pragma unroll
    for (int r = 0; r < 8; ++r) Bt[bnn][bk0 + 4 * r] = (_Float16)bReg[r];
    __syncthreads();
    if (k0 + 32 < 512) { // issue next K-tile loads; waits land at next store
      const int kn = k0 + 32;
#pragma unroll
      for (int r = 0; r < 16; ++r)
        aReg[r] = Vs[(size_t)(m0 + amr + 8 * r) * 512 + kn + akk];
#pragma unroll
      for (int r = 0; r < 8; ++r)
        bReg[r] = Pb[(size_t)(kn + bk0 + 4 * r) * 512 + n0 + bnn];
    }
    v16h a0 = frag_a(&As[wm + (lane & 15)][0], lane);
    v16h a1 = frag_a(&As[wm + 16 + (lane & 15)][0], lane);
    v16h b0 = frag_b(&Bt[wn + (lane & 15)][0], lane);
    v16h b1 = frag_b(&Bt[wn + 16 + (lane & 15)][0], lane);
    acc00 = __builtin_amdgcn_wmma_f32_16x16x32_f16(false, a0, false, b0, (short)0, acc00, false, false);
    acc01 = __builtin_amdgcn_wmma_f32_16x16x32_f16(false, a0, false, b1, (short)0, acc01, false, false);
    acc10 = __builtin_amdgcn_wmma_f32_16x16x32_f16(false, a1, false, b0, (short)0, acc10, false, false);
    acc11 = __builtin_amdgcn_wmma_f32_16x16x32_f16(false, a1, false, b1, (short)0, acc11, false, false);
    __syncthreads();
  }
  float* Sb = S + (size_t)b * NN2;
  const int hi = (lane >> 4) & 1;
  const int c0 = n0 + wn + (lane & 15);
  const int r0 = m0 + wm + 8 * hi;
#pragma unroll
  for (int r = 0; r < 8; ++r) {
    Sb[(size_t)(r0 + r) * 512 + c0]           = acc00[r];
    Sb[(size_t)(r0 + r) * 512 + c0 + 16]      = acc01[r];
    Sb[(size_t)(r0 + 16 + r) * 512 + c0]      = acc10[r];
    Sb[(size_t)(r0 + 16 + r) * 512 + c0 + 16] = acc11[r];
  }
}

// tmp[b](m, j) = sum_n cheb_k[n][m] * sAt[b][n][m] * X[b][n][j]   (JD = F*12)
// A-tile is the fused product, gathered transposed (contiguous in m).
template <int JD>
__launch_bounds__(256)
__global__ void gemm_tmp_kernel(const float* __restrict__ chebk, const float* __restrict__ sAt,
                                const float* __restrict__ X, float* __restrict__ TMP) {
  __shared__ _Float16 As[128][40];
  __shared__ _Float16 Bt[64][40];
  const int b = blockIdx.z;
  const int m0 = blockIdx.y * 128;
  const int j0 = blockIdx.x * 64;
  const int tid = threadIdx.x;
  const int lane = tid & 31, wave = tid >> 5;
  const int wm = (wave >> 1) * 32;
  const int wn = (wave & 1) * 32;
  const float* Ab = sAt + (size_t)b * NN2;
  const float* Xb = X + (size_t)b * Nn * JD;

  const int amr = tid & 127, ak0 = tid >> 7;  // A: k = ak0+2r (r<16)
  const int bnn = tid & 63, bk0 = tid >> 6;   // B: k = bk0+4r (r<8)
  const int jg = j0 + bnn;

  float aReg[16], bReg[8];
#pragma unroll
  for (int r = 0; r < 16; ++r) {
    size_t idx = (size_t)(ak0 + 2 * r) * 512 + m0 + amr;
    aReg[r] = chebk[idx] * Ab[idx];
  }
#pragma unroll
  for (int r = 0; r < 8; ++r)
    bReg[r] = (jg < JD) ? Xb[(size_t)(bk0 + 4 * r) * JD + jg] : 0.f;

  v8f acc00 = {}, acc01 = {}, acc10 = {}, acc11 = {};
  for (int k0 = 0; k0 < 512; k0 += 32) {
#pragma unroll
    for (int r = 0; r < 16; ++r) As[amr][ak0 + 2 * r] = (_Float16)aReg[r];
#pragma unroll
    for (int r = 0; r < 8; ++r) Bt[bnn][bk0 + 4 * r] = (_Float16)bReg[r];
    __syncthreads();
    if (k0 + 32 < 512) {
      const int kn = k0 + 32;
#pragma unroll
      for (int r = 0; r < 16; ++r) {
        size_t idx = (size_t)(kn + ak0 + 2 * r) * 512 + m0 + amr;
        aReg[r] = chebk[idx] * Ab[idx];
      }
#pragma unroll
      for (int r = 0; r < 8; ++r)
        bReg[r] = (jg < JD) ? Xb[(size_t)(kn + bk0 + 4 * r) * JD + jg] : 0.f;
    }
    v16h a0 = frag_a(&As[wm + (lane & 15)][0], lane);
    v16h a1 = frag_a(&As[wm + 16 + (lane & 15)][0], lane);
    v16h b0 = frag_b(&Bt[wn + (lane & 15)][0], lane);
    v16h b1 = frag_b(&Bt[wn + 16 + (lane & 15)][0], lane);
    acc00 = __builtin_amdgcn_wmma_f32_16x16x32_f16(false, a0, false, b0, (short)0, acc00, false, false);
    acc01 = __builtin_amdgcn_wmma_f32_16x16x32_f16(false, a0, false, b1, (short)0, acc01, false, false);
    acc10 = __builtin_amdgcn_wmma_f32_16x16x32_f16(false, a1, false, b0, (short)0, acc10, false, false);
    acc11 = __builtin_amdgcn_wmma_f32_16x16x32_f16(false, a1, false, b1, (short)0, acc11, false, false);
    __syncthreads();
  }
  const int hi = (lane >> 4) & 1;
  const int c0 = j0 + wn + (lane & 15);
  const int r0 = m0 + wm + 8 * hi;
  float* Tb = TMP + (size_t)b * Nn * JD;
#pragma unroll
  for (int r = 0; r < 8; ++r) {
    if (c0 < JD) {
      Tb[(size_t)(r0 + r) * JD + c0]      = acc00[r];
      Tb[(size_t)(r0 + 16 + r) * JD + c0] = acc10[r];
    }
    if (c0 + 16 < JD) {
      Tb[(size_t)(r0 + r) * JD + c0 + 16]      = acc01[r];
      Tb[(size_t)(r0 + 16 + r) * JD + c0 + 16] = acc11[r];
    }
  }
}

// ----------------------- Theta contraction + LN -----------------------------

// GCN layout: ((b*N+m)*T+t)*C + g   (channel-contiguous for the LN reduction)
template <int F>
__global__ void gcn_kernel(const float* __restrict__ TMP, const float* __restrict__ ThK,
                           float* __restrict__ GCN, int init) {
  int tid = blockIdx.x * blockDim.x + threadIdx.x;
  int g = tid & 63;
  int t = (tid >> 6) % Tt;
  int m = (tid / (64 * Tt)) & 511;
  int b = tid / (64 * Tt * Nn);
  const float* tp = TMP + ((size_t)(b * Nn + m)) * (F * Tt) + t;
  float acc = 0.f;
  for (int f = 0; f < F; ++f) acc += tp[f * Tt] * ThK[f * Cc + g];
  GCN[tid] = init ? acc : (GCN[tid] + acc);
}

// one wave per (b,n,t): relu(gcn)+res -> relu -> layernorm over C via shuffles
template <int F>
__global__ void ln_kernel(const float* __restrict__ GCN, const float* __restrict__ Xin,
                          const float* __restrict__ Wres, const float* __restrict__ bres,
                          const float* __restrict__ gamma, const float* __restrict__ beta,
                          float* __restrict__ Y) {
  int lane = threadIdx.x & 31;
  int wave = threadIdx.x >> 5;
  int id = blockIdx.x * 8 + wave; // id = (b*512+n)*12+t
  int t = id % Tt;
  int n = (id / Tt) & 511;
  int b = id / (Tt * Nn);
  const float* xp = Xin + ((size_t)(b * Nn + n)) * (F * Tt) + t;
  const float* gp = GCN + (size_t)id * Cc;

  float v[2], rsum = 0.f, rss = 0.f;
#pragma unroll
  for (int q = 0; q < 2; ++q) {
    int g = lane + 32 * q;
    float r = bres[g];
    for (int f = 0; f < F; ++f) r += xp[f * Tt] * Wres[g * F + f];
    float yv = fmaxf(r + fmaxf(gp[g], 0.f), 0.f);
    v[q] = yv;
    rsum += yv;
    rss += yv * yv;
  }
#pragma unroll
  for (int msk = 16; msk; msk >>= 1) {
    rsum += __shfl_xor(rsum, msk, 32);
    rss += __shfl_xor(rss, msk, 32);
  }
  float mu = rsum * (1.f / 64.f);
  float var = rss * (1.f / 64.f) - mu * mu;
  float rstd = rsqrtf(var + 1e-5f);
#pragma unroll
  for (int q = 0; q < 2; ++q) {
    int g = lane + 32 * q;
    Y[(((size_t)(b * Nn + n)) * Cc + g) * Tt + t] = (v[q] - mu) * rstd * gamma[g] + beta[g];
  }
}

// ------------------------- conv head, GRU, output ---------------------------

__global__ void gout_kernel(const float* __restrict__ Y, const float* __restrict__ Wfc,
                            const float* __restrict__ bfc, float* __restrict__ G) {
  int tid = blockIdx.x * blockDim.x + threadIdx.x; // (b*12+o)*512+n
  int n = tid & 511;
  int o = (tid >> 9) % Tt;
  int b = tid / (Tt * Nn);
  const float* yp = Y + ((size_t)(b * Nn + n)) * Cc * Tt;
  const float* wp = Wfc + o * Tt * Cc;
  float s = bfc[o];
  for (int t = 0; t < Tt; ++t)
    for (int f = 0; f < Cc; ++f) s += yp[f * Tt + t] * wp[t * Cc + f];
  G[tid] = s;
}

// g = sigmoid([x_t, h] @ Wg1 + bg1); GB layout b*65536 + n*128 + c2
__global__ void gru_gate_kernel(const float* __restrict__ G, const float* __restrict__ H,
                                const float* __restrict__ Wg1, const float* __restrict__ bg1,
                                float* __restrict__ GB, int t) {
  int tid = blockIdx.x * blockDim.x + threadIdx.x;
  int c2 = tid & 127;
  int n = (tid >> 7) & 511;
  int b = tid >> 16;
  float xt = G[((size_t)b * Tt + t) * Nn + n];
  const float* hp = H + (size_t)b * (Nn * Cc) + n * Cc;
  float s = bg1[c2] + xt * Wg1[c2];
  for (int j = 0; j < Cc; ++j) s += hp[j] * Wg1[(j + 1) * 128 + c2];
  GB[tid] = 1.f / (1.f + __expf(-s));
}

// reference quirk reproduced exactly: r = g_flat[:, :N*H], u = g_flat[:, N*H:]
// (flat indexing across nodes), rh[i] = g_flat[i]*h_flat[i], u[i] = g_flat[NH+i]
__global__ void gru_update_kernel(const float* __restrict__ G, const float* __restrict__ GB,
                                  float* __restrict__ H, const float* __restrict__ Wg2,
                                  const float* __restrict__ bg2, int t) {
  __shared__ float rh[64];
  int b = blockIdx.x >> 9, n = blockIdx.x & 511;
  int c = threadIdx.x; // 0..63
  int i = n * Cc + c;  // flat index within batch
  const float* gb = GB + (size_t)b * 65536;
  float* hp = H + (size_t)b * (Nn * Cc);
  float hv = hp[i];
  rh[c] = gb[i] * hv;
  __syncthreads();
  float xt = G[((size_t)b * Tt + t) * Nn + n];
  float s = bg2[c] + xt * Wg2[c];
  for (int j = 0; j < Cc; ++j) s += rh[j] * Wg2[(j + 1) * Cc + c];
  float cv = tanhf(s);
  float u = gb[32768 + i];
  hp[i] = u * hv + (1.f - u) * cv;
}

__global__ void final_kernel(const float* __restrict__ H, const float* __restrict__ x,
                             const float* __restrict__ Wfl, const float* __restrict__ bfl,
                             const float* __restrict__ Ww, const float* __restrict__ bw,
                             const float* __restrict__ Wd, const float* __restrict__ bd,
                             float* __restrict__ out) {
  int tid = blockIdx.x * blockDim.x + threadIdx.x; // (b*512+n)*12+tt
  int tt = tid % Tt;
  int bn = tid / Tt;
  const float* hp = H + (size_t)bn * Cc;
  const float* xp = x + (size_t)bn * 36;
  float s = bfl[tt] + bw[tt] + bd[tt];
  for (int g = 0; g < Cc; ++g) s += hp[g] * Wfl[g * Tt + tt];
#pragma unroll
  for (int j = 0; j < Tt; ++j) s += xp[j] * Ww[j * Tt + tt] + xp[12 + j] * Wd[j * Tt + tt];
  out[tid] = s;
}

// ------------------------------- host side ----------------------------------

struct BlockParams {
  const float *W1, *W2, *W3, *bs, *Vs, *Theta, *Wres, *bres, *gamma, *beta;
};

template <int F>
static void run_block(const float* Xin, const BlockParams& p, const float* cheb, float* Y,
                      float* L, float* R, float* P, float* S, float* TMP, float* GCN,
                      hipStream_t stream) {
  lr_kernel<F><<<64, 256, 0, stream>>>(Xin, p.W1, p.W2, p.W3, L, R);
  prod_sig_kernel<<<32768, 256, 0, stream>>>(L, R, p.bs, P);
  gemm_s_kernel<<<dim3(8, 4, Bb), 256, 0, stream>>>(p.Vs, P, S);
  softmax_col_kernel<<<64, 256, 0, stream>>>(S); // S -> sAt in place
  constexpr int JD = F * Tt;
  for (int k = 0; k < 3; ++k) {
    gemm_tmp_kernel<JD><<<dim3((JD + 63) / 64, 4, Bb), 256, 0, stream>>>(
        cheb + (size_t)k * NN2, S, Xin, TMP);
    gcn_kernel<F><<<49152, 256, 0, stream>>>(TMP, p.Theta + (size_t)k * F * Cc, GCN,
                                             (k == 0) ? 1 : 0);
  }
  ln_kernel<F><<<24576, 256, 0, stream>>>(GCN, Xin, p.Wres, p.bres, p.gamma, p.beta, Y);
}

// workspace layout (bytes); total ~271 MB
static const size_t OFF_X0 = 0;                   // 786432
static const size_t OFF_Y0 = OFF_X0 + 786432;     // 50331648
static const size_t OFF_Y1 = OFF_Y0 + 50331648;   // 50331648
static const size_t OFF_L = OFF_Y1 + 50331648;    // 786432
static const size_t OFF_R = OFF_L + 786432;       // 786432
static const size_t OFF_P = OFF_R + 786432;       // 33554432
static const size_t OFF_S = OFF_P + 33554432;     // 33554432
static const size_t OFF_TMP = OFF_S + 33554432;   // 50331648
static const size_t OFF_GCN = OFF_TMP + 50331648; // 50331648
static const size_t OFF_GOUT = OFF_GCN + 50331648;// 786432
static const size_t OFF_GBUF = OFF_GOUT + 786432; // 8388608
static const size_t OFF_H = OFF_GBUF + 8388608;   // 4194304

extern "C" void kernel_launch(void* const* d_in, const int* in_sizes, int n_in,
                              void* d_out, int out_size, void* d_ws, size_t ws_size,
                              hipStream_t stream) {
  (void)in_sizes; (void)n_in; (void)out_size; (void)ws_size;
  const float* x = (const float*)d_in[0];
  BlockParams bp0 = {(const float*)d_in[1], (const float*)d_in[2], (const float*)d_in[3],
                     (const float*)d_in[4], (const float*)d_in[5], (const float*)d_in[6],
                     (const float*)d_in[7], (const float*)d_in[8], (const float*)d_in[9],
                     (const float*)d_in[10]};
  BlockParams bp1 = {(const float*)d_in[11], (const float*)d_in[12], (const float*)d_in[13],
                     (const float*)d_in[14], (const float*)d_in[15], (const float*)d_in[16],
                     (const float*)d_in[17], (const float*)d_in[18], (const float*)d_in[19],
                     (const float*)d_in[20]};
  const float* Wfc = (const float*)d_in[21];
  const float* bfc = (const float*)d_in[22];
  const float* Wg1 = (const float*)d_in[23];
  const float* bg1 = (const float*)d_in[24];
  const float* Wg2 = (const float*)d_in[25];
  const float* bg2 = (const float*)d_in[26];
  const float* Wfl = (const float*)d_in[27];
  const float* bfl = (const float*)d_in[28];
  const float* Ww = (const float*)d_in[29];
  const float* bw = (const float*)d_in[30];
  const float* Wd = (const float*)d_in[31];
  const float* bd = (const float*)d_in[32];
  const float* cheb = (const float*)d_in[33];

  char* w = (char*)d_ws;
  float* X0 = (float*)(w + OFF_X0);
  float* Y0 = (float*)(w + OFF_Y0);
  float* Y1 = (float*)(w + OFF_Y1);
  float* L = (float*)(w + OFF_L);
  float* R = (float*)(w + OFF_R);
  float* P = (float*)(w + OFF_P);
  float* S = (float*)(w + OFF_S);
  float* TMP = (float*)(w + OFF_TMP);
  float* GCN = (float*)(w + OFF_GCN);
  float* GOUT = (float*)(w + OFF_GOUT);
  float* GBUF = (float*)(w + OFF_GBUF);
  float* Hst = (float*)(w + OFF_H);

  extract_kernel<<<768, 256, 0, stream>>>(x, X0);
  run_block<1>(X0, bp0, cheb, Y0, L, R, P, S, TMP, GCN, stream);
  run_block<64>(Y0, bp1, cheb, Y1, L, R, P, S, TMP, GCN, stream);
  gout_kernel<<<768, 256, 0, stream>>>(Y1, Wfc, bfc, GOUT);
  zero_kernel<<<4096, 256, 0, stream>>>(Hst, Bb * Nn * Cc);
  for (int t = 0; t < Tt; ++t) {
    gru_gate_kernel<<<8192, 256, 0, stream>>>(GOUT, Hst, Wg1, bg1, GBUF, t);
    gru_update_kernel<<<16384, 64, 0, stream>>>(GOUT, GBUF, Hst, Wg2, bg2, t);
  }
  final_kernel<<<768, 256, 0, stream>>>(Hst, x, Wfl, bfl, Ww, bw, Wd, bd, (float*)d_out);
}
